// TemporalGNN_3161095930178
// MI455X (gfx1250) — matrix operand order repaired and checked by reading, not compiled, not run on hardware
//
#include <hip/hip_runtime.h>
#include <hip/hip_bf16.h>

typedef __attribute__((ext_vector_type(16))) __bf16 v16bf;
typedef __attribute__((ext_vector_type(8)))  float  v8f;

#define Hh 128
#define LDSP 130   // LDS row pitch in ushorts: 260B -> conflict-friendly, even-K reads 4B aligned

__device__ __forceinline__ unsigned short f2bf(float f) {
  unsigned u = __builtin_bit_cast(unsigned, f);
  unsigned r = u + 0x7FFFu + ((u >> 16) & 1u);   // RNE
  return (unsigned short)(r >> 16);
}
__device__ __forceinline__ unsigned pk2(float a, float b) {
  return (unsigned)f2bf(a) | ((unsigned)f2bf(b) << 16);
}

union AFrag { v16bf v; unsigned u[8]; };

__device__ __forceinline__ void load_bfrag(AFrag& f, const unsigned* __restrict__ base) {
  uint4 p0 = ((const uint4*)base)[0];
  uint4 p1 = ((const uint4*)base)[1];
  f.u[0] = p0.x; f.u[1] = p0.y; f.u[2] = p0.z; f.u[3] = p0.w;
  f.u[4] = p1.x; f.u[5] = p1.y; f.u[6] = p1.z; f.u[7] = p1.w;
}

// ---------------- degree / normalization ----------------
__global__ void k_deg_init(float* __restrict__ deg, int n) {
  int i = blockIdx.x * blockDim.x + threadIdx.x;
  if (i < n) deg[i] = 1.0f;                         // self-loop contributes 1
}
__global__ void k_deg_edges(const long long* __restrict__ ei, float* __restrict__ deg, long long E) {
  long long i = (long long)blockIdx.x * blockDim.x + threadIdx.x;
  if (i < E) atomicAdd(deg + (int)ei[E + i], 1.0f); // dst row of edge_index
}
__global__ void k_dinv(const float* __restrict__ deg, float* __restrict__ dinv, int n) {
  int i = blockIdx.x * blockDim.x + threadIdx.x;
  if (i < n) dinv[i] = rsqrtf(deg[i]);
}

// -------- pre-pack weights into WMMA B-fragment layout (bf16 pairs) --------
// W_gcn [128][128] row-major, B[k][n]=Wg[k][n]. out[8 tiles][4 ksteps][32 lanes][8 dwords]
__global__ void k_pack_wg(const float* __restrict__ Wg, unsigned* __restrict__ out) {
  int t = blockIdx.x * blockDim.x + threadIdx.x;
  if (t >= 8 * 4 * 32) return;
  int lane = t & 31, kk = (t >> 5) & 3, c = t >> 7;
  int lo = lane & 15, hi = lane >> 4;
  int n = c * 16 + lo;
  unsigned* o = out + t * 8;
#pragma unroll
  for (int v = 0; v < 8; ++v) {
    int K = kk * 32 + ((v < 4) ? 2 * v : 2 * v + 8) + 8 * hi;
    o[v] = pk2(Wg[K * Hh + n], Wg[(K + 1) * Hh + n]);
  }
}
// w_ih [384][128] row-major, B[k][m]=wih[m][k]. out[24 tiles][4][32][8]
__global__ void k_pack_wih(const float* __restrict__ wih, unsigned* __restrict__ out) {
  int t = blockIdx.x * blockDim.x + threadIdx.x;
  if (t >= 24 * 4 * 32) return;
  int lane = t & 31, kk = (t >> 5) & 3, c = t >> 7;     // c in 0..23 spans the 3 gates
  int lo = lane & 15, hi = lane >> 4;
  int m = c * 16 + lo;
  unsigned* o = out + t * 8;
#pragma unroll
  for (int v = 0; v < 8; ++v) {
    int K = kk * 32 + ((v < 4) ? 2 * v : 2 * v + 8) + 8 * hi;
    o[v] = pk2(wih[m * Hh + K], wih[m * Hh + K + 1]);
  }
}

// ---------------- GEMM1: xw = x @ W_gcn (bf16 WMMA, f32 accum) ----------------
__global__ __launch_bounds__(256) void k_gemm_gcn(const float* __restrict__ x,
                                                  const unsigned* __restrict__ wgp,
                                                  float* __restrict__ xw, int N) {
  extern __shared__ char smem[];
  unsigned short* a_lds = (unsigned short*)smem;                  // [128][LDSP]

  const int tid  = threadIdx.x;
  const int wave = tid >> 5;
  const int lane = tid & 31;
  const int lo   = lane & 15, hi = lane >> 4;
  const long long rowBlock = (long long)blockIdx.x * 128;
  const bool full = (rowBlock + 128 <= (long long)N);

  for (int idx = tid; idx < 128 * 32; idx += 256) {
    int r = idx >> 5, q = idx & 31;
    long long g = rowBlock + r;
    float4 v = make_float4(0.f, 0.f, 0.f, 0.f);
    if (g < N) v = ((const float4*)(x + g * Hh))[q];
    unsigned short* p = a_lds + r * LDSP + q * 4;
    p[0] = f2bf(v.x); p[1] = f2bf(v.y); p[2] = f2bf(v.z); p[3] = f2bf(v.w);
  }
  __syncthreads();

  AFrag af[4];
  const int arow = wave * 16 + lo;
#pragma unroll
  for (int kk = 0; kk < 4; ++kk)
#pragma unroll
    for (int v = 0; v < 8; ++v) {
      int K = kk * 32 + ((v < 4) ? 2 * v : 2 * v + 8) + 8 * hi;   // ISA 16-bit A layout
      af[kk].u[v] = *(const unsigned*)(a_lds + arow * LDSP + K);
    }

  float* ob = xw + (size_t)rowBlock * Hh;
  const int rbase = wave * 16 + 8 * hi;                           // C layout: row=v+8*hi, col=lo
#pragma unroll
  for (int c = 0; c < 8; ++c) {
    v8f acc = {};
#pragma unroll
    for (int kk = 0; kk < 4; ++kk) {
      AFrag bf;
      load_bfrag(bf, wgp + (((c << 2) + kk) << 8) + (lane << 3));
      acc = __builtin_amdgcn_wmma_f32_16x16x32_bf16(false, af[kk].v, false, bf.v,
                                                    (short)0, acc, false, false);
    }
    const int coff = c * 16 + lo;
    if (full) {
#pragma unroll
      for (int vv = 0; vv < 8; ++vv) ob[(rbase + vv) * Hh + coff] = acc[vv];
    } else {
#pragma unroll
      for (int vv = 0; vv < 8; ++vv)
        if (rowBlock + rbase + vv < (long long)N) ob[(rbase + vv) * Hh + coff] = acc[vv];
    }
  }
}

// ---------------- self-loop init: agg = xw * dinv^2 ----------------
__global__ void k_selfloop(const float* __restrict__ xw, const float* __restrict__ dinv,
                           float* __restrict__ agg, long long n32) {
  long long g = (long long)blockIdx.x * blockDim.x + threadIdx.x;
  if (g >= n32) return;
  int node = (int)(g >> 5), q = (int)(g & 31);
  float di = dinv[node], s = di * di;
  float4 v = ((const float4*)(xw + (size_t)node * Hh))[q];
  float4 o; o.x = v.x * s; o.y = v.y * s; o.z = v.z * s; o.w = v.w * s;
  ((float4*)(agg + (size_t)node * Hh))[q] = o;
}

// ---------------- edge scatter: agg[dst] += xw[src]*norm (L2-resident) ----------------
__global__ void k_scatter(const long long* __restrict__ ei, const float* __restrict__ xw,
                          const float* __restrict__ dinv, float* __restrict__ agg, long long E) {
  long long g = (long long)blockIdx.x * blockDim.x + threadIdx.x;
  if (g >= E * 32) return;
  long long e = g >> 5; int q = (int)(g & 31);
  int s = (int)ei[e], d = (int)ei[E + e];
  float nrm = dinv[s] * dinv[d];
  float4 v = ((const float4*)(xw + (size_t)s * Hh))[q];
  float* ap = agg + (size_t)d * Hh + q * 4;
  atomicAdd(ap + 0, v.x * nrm);
  atomicAdd(ap + 1, v.y * nrm);
  atomicAdd(ap + 2, v.z * nrm);
  atomicAdd(ap + 3, v.w * nrm);
}

// ---------------- fused: h=relu(agg+b); gi=h@w_ih.T; gates (gh=b_hh); head ----------------
__global__ __launch_bounds__(256) void k_gru_head(const float* __restrict__ agg,
                                                  const float* __restrict__ b_gcn,
                                                  const unsigned* __restrict__ wihp,
                                                  const float* __restrict__ b_ih,
                                                  const float* __restrict__ b_hh,
                                                  const float* __restrict__ W_lin,
                                                  const float* __restrict__ b_lin,
                                                  float* __restrict__ out, int N) {
  extern __shared__ char smem[];
  unsigned short* h_lds = (unsigned short*)smem;                  // [128][LDSP]
  const int tid  = threadIdx.x;
  const int wave = tid >> 5;
  const int lane = tid & 31;
  const int lo   = lane & 15, hi = lane >> 4;
  const long long rowBlock = (long long)blockIdx.x * 128;
  const bool full = (rowBlock + 128 <= (long long)N);

  for (int idx = tid; idx < 128 * 32; idx += 256) {
    int r = idx >> 5, q = idx & 31;
    long long g = rowBlock + r;
    float4 v = make_float4(0.f, 0.f, 0.f, 0.f);
    if (g < N) v = ((const float4*)(agg + g * Hh))[q];
    float4 b = ((const float4*)b_gcn)[q];
    unsigned short* p = h_lds + r * LDSP + q * 4;
    p[0] = f2bf(fmaxf(v.x + b.x, 0.f)); p[1] = f2bf(fmaxf(v.y + b.y, 0.f));
    p[2] = f2bf(fmaxf(v.z + b.z, 0.f)); p[3] = f2bf(fmaxf(v.w + b.w, 0.f));
  }
  __syncthreads();

  AFrag af[4];
  const int arow = wave * 16 + lo;
#pragma unroll
  for (int kk = 0; kk < 4; ++kk)
#pragma unroll
    for (int v = 0; v < 8; ++v) {
      int K = kk * 32 + ((v < 4) ? 2 * v : 2 * v + 8) + 8 * hi;
      af[kk].u[v] = *(const unsigned*)(h_lds + arow * LDSP + K);
    }

  float acc_head[8] = {0.f, 0.f, 0.f, 0.f, 0.f, 0.f, 0.f, 0.f};
#pragma unroll
  for (int c = 0; c < 8; ++c) {
    v8f ar = {}, az = {}, an = {};
    const int j = c * 16 + lo;                                    // gate column
#pragma unroll
    for (int kk = 0; kk < 4; ++kk) {
      AFrag br, bz, bn;
      load_bfrag(br, wihp + ((( 0 + c) * 4 + kk) << 8) + (lane << 3));
      load_bfrag(bz, wihp + ((( 8 + c) * 4 + kk) << 8) + (lane << 3));
      load_bfrag(bn, wihp + (((16 + c) * 4 + kk) << 8) + (lane << 3));
      ar = __builtin_amdgcn_wmma_f32_16x16x32_bf16(false, af[kk].v, false, br.v, (short)0, ar, false, false);
      az = __builtin_amdgcn_wmma_f32_16x16x32_bf16(false, af[kk].v, false, bz.v, (short)0, az, false, false);
      an = __builtin_amdgcn_wmma_f32_16x16x32_bf16(false, af[kk].v, false, bn.v, (short)0, an, false, false);
    }
    // gates: gh = b_hh exactly (h0 == 0)
    float bir  = b_ih[j] + b_hh[j];
    float biz  = b_ih[128 + j] + b_hh[128 + j];
    float bin_ = b_ih[256 + j];
    float bhn  = b_hh[256 + j];
    float wl   = W_lin[j];
#pragma unroll
    for (int v = 0; v < 8; ++v) {
      float r  = 1.f / (1.f + __expf(-(ar[v] + bir)));
      float z  = 1.f / (1.f + __expf(-(az[v] + biz)));
      float nn = tanhf(an[v] + bin_ + r * bhn);
      float hn = (1.f - z) * nn;                                   // + z*h0, h0 == 0
      acc_head[v] += hn * wl;
    }
  }
  const float bl = b_lin[0];
#pragma unroll
  for (int v = 0; v < 8; ++v) {
    float s = acc_head[v];
    s += __shfl_xor(s, 1, 32);
    s += __shfl_xor(s, 2, 32);
    s += __shfl_xor(s, 4, 32);
    s += __shfl_xor(s, 8, 32);
    int r = wave * 16 + v + 8 * hi;
    if (lo == 0 && (full || rowBlock + r < (long long)N)) out[rowBlock + r] = s + bl;
  }
}

extern "C" void kernel_launch(void* const* d_in, const int* in_sizes, int n_in,
                              void* d_out, int out_size, void* d_ws, size_t ws_size,
                              hipStream_t stream) {
  const float*     x    = (const float*)d_in[0];
  const long long* ei   = (const long long*)d_in[1];
  const float*     Wg   = (const float*)d_in[2];
  const float*     bg   = (const float*)d_in[3];
  const float*     wih  = (const float*)d_in[4];
  // d_in[5] = w_hh: unused (h0 == 0 -> gh = b_hh)
  const float*     bih  = (const float*)d_in[6];
  const float*     bhh  = (const float*)d_in[7];
  const float*     Wlin = (const float*)d_in[8];
  const float*     blin = (const float*)d_in[9];
  float*           out  = (float*)d_out;

  const long long N = in_sizes[0] / Hh;
  const long long E = in_sizes[1] / 2;

  char* ws = (char*)d_ws;
  size_t o = 0;
  auto alignup = [](size_t v) { return (v + 255) & ~(size_t)255; };
  float*    deg  = (float*)(ws + o);    o = alignup(o + (size_t)N * 4);
  float*    dinv = (float*)(ws + o);    o = alignup(o + (size_t)N * 4);
  float*    xw   = (float*)(ws + o);    o = alignup(o + (size_t)N * Hh * 4);
  float*    agg  = (float*)(ws + o);    o = alignup(o + (size_t)N * Hh * 4);
  unsigned* wgp  = (unsigned*)(ws + o); o = alignup(o + (size_t)8  * 4 * 32 * 8 * 4);
  unsigned* wihp = (unsigned*)(ws + o); o = alignup(o + (size_t)24 * 4 * 32 * 8 * 4);

  const int T = 256;
  const int nblk = (int)((N + 127) / 128);
  const size_t ldsA = (size_t)128 * LDSP * sizeof(unsigned short);  // 33,280 B

  k_pack_wg  <<<(8 * 4 * 32 + T - 1) / T, T, 0, stream>>>(Wg, wgp);
  k_pack_wih <<<(24 * 4 * 32 + T - 1) / T, T, 0, stream>>>(wih, wihp);
  k_deg_init <<<(int)((N + T - 1) / T), T, 0, stream>>>(deg, (int)N);
  k_deg_edges<<<(int)((E + T - 1) / T), T, 0, stream>>>(ei, deg, E);
  k_dinv     <<<(int)((N + T - 1) / T), T, 0, stream>>>(deg, dinv, (int)N);
  k_gemm_gcn <<<nblk, T, ldsA, stream>>>(x, wgp, xw, (int)N);
  k_selfloop <<<(int)((N * 32 + T - 1) / T), T, 0, stream>>>(xw, dinv, agg, N * 32);
  k_scatter  <<<(int)((E * 32 + T - 1) / T), T, 0, stream>>>(ei, xw, dinv, agg, E);
  k_gru_head <<<nblk, T, ldsA, stream>>>(agg, bg, wihp, bih, bhh, Wlin, blin, out, (int)N);
}
